// attn_60327110639979
// MI455X (gfx1250) — compile-verified, matrix operand order checked by reading, not run on hardware
//
#include <hip/hip_runtime.h>

typedef __attribute__((ext_vector_type(16))) _Float16 v16h;
typedef __attribute__((ext_vector_type(8)))  float    v8f;
typedef __attribute__((ext_vector_type(8)))  int      v8i;

#define TT     4
#define BB     4
#define CC     384
#define NN     4096
#define NHEADS 6
#define DH     64
#define NIMG   16
#define HP     66
#define PADAREA (HP*HP)

__device__ __forceinline__ int kbase16(int i, int lane) {
  int b = (i < 4) ? (i * 2) : (16 + (i - 4) * 2);
  return b + ((lane & 16) ? 8 : 0);
}
__device__ __forceinline__ int kbase8(int i, int lane) {
  int b = ((i & 1) ? 4 : 0) + ((i & 2) ? 16 : 0) + ((i & 4) ? 32 : 0);
  return b + ((lane & 16) ? 8 : 0);
}
union FragH { v16h h; unsigned int u[8]; };
union FragB { v8i  v; unsigned int u[8]; };

__global__ void cvt_w_kernel(const float* __restrict__ w1, const float* __restrict__ pw,
                             _Float16* __restrict__ wf16) {
  int idx = blockIdx.x * 256 + threadIdx.x;
  const int per = 4 * CC * CC;
  float v = (idx < per) ? w1[idx] : pw[idx - per];
  wf16[idx] = (_Float16)v;
}

__global__ void bn_prep_kernel(const float* b1w, const float* b1b, const float* b1m, const float* b1v,
                               const float* b2w, const float* b2b, const float* b2m, const float* b2v,
                               const float* b3w, const float* b3b, const float* b3m, const float* b3v,
                               float* __restrict__ coef) {
  int idx = blockIdx.x * 256 + threadIdx.x;
  if (idx >= 4 * CC) return;
  float s1 = b1w[idx] * rsqrtf(b1v[idx] + 1e-5f);
  float t1 = b1b[idx] - b1m[idx] * s1;
  float s2 = b2w[idx] * rsqrtf(b2v[idx] + 1e-5f);
  float t2 = b2b[idx] - b2m[idx] * s2;
  float s3 = b3w[idx] * rsqrtf(b3v[idx] + 1e-5f);
  float t3 = b3b[idx] - b3m[idx] * s3;
  int i = idx / CC, c = idx % CC;
  coef[(i * 4 + 0) * CC + c] = s1;
  coef[(i * 4 + 1) * CC + c] = t1;
  coef[(i * 4 + 2) * CC + c] = s2 * s3;
  coef[(i * 4 + 3) * CC + c] = t2 * s3 + t3;
}

__global__ void alif_kernel(const float* __restrict__ in, const float* __restrict__ vth, int vidx,
                            _Float16* __restrict__ out16, unsigned char* __restrict__ out8, int mode) {
  size_t idx = (size_t)blockIdx.x * 256 + threadIdx.x;
  const size_t tot = (size_t)BB * CC * NN;
  if (idx >= tot) return;
  int b   = (int)(idx / ((size_t)CC * NN));
  int rem = (int)(idx % ((size_t)CC * NN));
  int c = rem / NN;
  int n = rem % NN;
  float th = vth[vidx];
  float v = 0.f;
#pragma unroll
  for (int t = 0; t < TT; ++t) {
    float x = in[(size_t)((t * BB + b) * CC + c) * NN + n];
    v = 0.5f * (v + x);
    float s = (v - th) >= 0.f ? 1.f : 0.f;
    int tb = t * BB + b;
    if (mode == 0) {
      out16[((size_t)tb * NN + n) * CC + c] = (_Float16)s;
    } else {
      int h = c >> 6, dd = c & 63;
      if (mode == 1) out16[(((size_t)tb * NHEADS + h) * NN + n) * DH + dd] = (_Float16)s;
      else           out8 [(((size_t)tb * NHEADS + h) * DH + dd) * NN + n] = (unsigned char)s;
    }
  }
}

__global__ void padfill_kernel(const float* __restrict__ shift, _Float16* __restrict__ ypad) {
  size_t idx = (size_t)blockIdx.x * 256 + threadIdx.x;
  const size_t tot = (size_t)NIMG * CC * PADAREA;
  if (idx >= tot) return;
  int c = (int)((idx / PADAREA) % CC);
  ypad[idx] = (_Float16)shift[c];
}

__global__ void conv1x1_wmma_kernel(const _Float16* __restrict__ Wf16,
                                    const _Float16* __restrict__ xT,
                                    const float* __restrict__ scale,
                                    const float* __restrict__ shift,
                                    _Float16* __restrict__ outPad,
                                    float* __restrict__ outF32,
                                    int mode) {
  __shared__ unsigned int wtile[1024];
  const int lane = threadIdx.x & 31;
  const int wave = threadIdx.x >> 5;
  const int img  = blockIdx.z;
  const int o0   = blockIdx.y << 6;
  const int n0   = (blockIdx.x << 7) + (wave << 4);
  const unsigned int* xw = (const unsigned int*)(xT + (size_t)img * NN * CC);
  const unsigned int* ww = (const unsigned int*)Wf16;

  const int m  = threadIdx.x >> 2;
  const int wg = (threadIdx.x & 3) << 2;
  typedef __attribute__((address_space(3))) unsigned int lds_u32;
  const unsigned ldsoff =
      (unsigned)(size_t)((lds_u32*)wtile) + (unsigned)(((m << 4) + wg) << 2);
  const unsigned int* wsrc = ww + (size_t)(o0 + m) * 192 + wg;
  const unsigned int* xrow = xw + (size_t)(n0 + (lane & 15)) * 192;

  v8f acc0 = {}, acc1 = {}, acc2 = {}, acc3 = {};
#pragma unroll
  for (int ks = 0; ks < 12; ++ks) {
    __syncthreads();
    {
      const unsigned int* gsrc = wsrc + ks * 16;
      asm volatile("global_load_async_to_lds_b128 %0, %1, off"
                   :: "v"(ldsoff), "v"(gsrc) : "memory");
      asm volatile("s_wait_asynccnt 0" ::: "memory");
    }
    __syncthreads();

    FragH b;
#pragma unroll
    for (int i = 0; i < 8; ++i) {
      int kb = kbase16(i, lane) >> 1;
      b.u[i] = xrow[ks * 16 + kb];
    }
    if (ks + 1 < 12)
      __builtin_prefetch(xrow + (ks + 1) * 16, 0, 1);

    FragH a0, a1, a2, a3;
#pragma unroll
    for (int i = 0; i < 8; ++i) {
      int kb = kbase16(i, lane) >> 1;
      int row = ((lane & 15) << 4) + kb;
      a0.u[i] = wtile[row];
      a1.u[i] = wtile[row + (16 << 4)];
      a2.u[i] = wtile[row + (32 << 4)];
      a3.u[i] = wtile[row + (48 << 4)];
    }
    acc0 = __builtin_amdgcn_wmma_f32_16x16x32_f16(false, a0.h, false, b.h,
                                                  (short)0, acc0, false, false);
    acc1 = __builtin_amdgcn_wmma_f32_16x16x32_f16(false, a1.h, false, b.h,
                                                  (short)0, acc1, false, false);
    acc2 = __builtin_amdgcn_wmma_f32_16x16x32_f16(false, a2.h, false, b.h,
                                                  (short)0, acc2, false, false);
    acc3 = __builtin_amdgcn_wmma_f32_16x16x32_f16(false, a3.h, false, b.h,
                                                  (short)0, acc3, false, false);
  }

  const int col = lane & 15;
  const int hi  = (lane >> 4) & 1;
  const int n   = n0 + col;
  auto epi = [&](const v8f& a, int ot) {
#pragma unroll
    for (int r = 0; r < 8; ++r) {
      int o = o0 + (ot << 4) + r + (hi << 3);
      float v = a[r] * scale[o] + shift[o];
      if (mode == 0) {
        int hh = n >> 6, wc = n & 63;
        outPad[(size_t)img * CC * PADAREA + (size_t)o * PADAREA
               + (size_t)(hh + 1) * HP + (wc + 1)] = (_Float16)v;
      } else {
        outF32[(size_t)img * CC * NN + (size_t)o * NN + n] = v;
      }
    }
  };
  epi(acc0, 0); epi(acc1, 1); epi(acc2, 2); epi(acc3, 3);
}

__global__ void dwconv_kernel(const _Float16* __restrict__ ypad,
                              const float* __restrict__ dw,
                              _Float16* __restrict__ y2T) {
  size_t idx = (size_t)blockIdx.x * 256 + threadIdx.x;
  const size_t tot = (size_t)NIMG * CC * NN;
  if (idx >= tot) return;
  int n   = (int)(idx % NN);
  int c   = (int)((idx / NN) % CC);
  int img = (int)(idx / ((size_t)NN * CC));
  int hh = n >> 6, wc = n & 63;
  const _Float16* src = ypad + (size_t)img * CC * PADAREA + (size_t)c * PADAREA;
  const float* wgt = dw + c * 9;
  float a = 0.f;
#pragma unroll
  for (int dy = 0; dy < 3; ++dy)
#pragma unroll
    for (int dx = 0; dx < 3; ++dx)
      a += (float)src[(hh + dy) * HP + (wc + dx)] * wgt[dy * 3 + dx];
  y2T[((size_t)img * NN + n) * CC + c] = (_Float16)a;
}

__global__ void kv_wmma_kernel(const unsigned char* __restrict__ kT,
                               const unsigned char* __restrict__ vT,
                               _Float16* __restrict__ kvT) {
  const int lane = threadIdx.x & 31;
  const int wave = threadIdx.x >> 5;
  const int bh = blockIdx.x;
  const int d0 = (wave >> 2) << 4;
  const int e0 = (wave & 3) << 4;
  const unsigned char* kb = kT + (size_t)bh * DH * NN;
  const unsigned char* vb = vT + (size_t)bh * DH * NN;
  v8i acc = {};
  for (int ks = 0; ks < 64; ++ks) {
    int nb = ks << 6;
    FragB a, b;
#pragma unroll
    for (int i = 0; i < 8; ++i) {
      int o = nb + kbase8(i, lane);
      a.u[i] = *(const unsigned int*)(kb + (size_t)(d0 + (lane & 15)) * NN + o);
      b.u[i] = *(const unsigned int*)(vb + (size_t)(e0 + (lane & 15)) * NN + o);
    }
    acc = __builtin_amdgcn_wmma_i32_16x16x64_iu8(false, a.v, false, b.v, acc, false, false);
  }
  const int col = lane & 15;
  const int hi  = (lane >> 4) & 1;
#pragma unroll
  for (int r = 0; r < 8; ++r) {
    int d = d0 + r + (hi << 3);
    int e = e0 + col;
    kvT[((size_t)bh * DH + e) * DH + d] = (_Float16)(float)acc[r];
  }
}

__global__ void qkv_attn_kernel(const _Float16* __restrict__ q,
                                const _Float16* __restrict__ kvT,
                                float* __restrict__ ua) {
  const int lane = threadIdx.x & 31;
  const int wave = threadIdx.x >> 5;
  const int bh = blockIdx.y;
  int id = (blockIdx.x << 3) + wave;
  int n0 = (id >> 2) << 4;
  int e0 = (id & 3) << 4;
  const unsigned int* qw = (const unsigned int*)(q + (size_t)bh * NN * DH);
  const unsigned int* kw = (const unsigned int*)(kvT + (size_t)bh * DH * DH);
  v8f acc = {};
#pragma unroll
  for (int ks = 0; ks < 2; ++ks) {
    FragH a, b;
#pragma unroll
    for (int i = 0; i < 8; ++i) {
      int kb = (kbase16(i, lane) + (ks << 5)) >> 1;
      a.u[i] = qw[(size_t)(n0 + (lane & 15)) * 32 + kb];
      b.u[i] = kw[(size_t)(e0 + (lane & 15)) * 32 + kb];
    }
    acc = __builtin_amdgcn_wmma_f32_16x16x32_f16(false, a.h, false, b.h,
                                                 (short)0, acc, false, false);
  }
  const int col = lane & 15;
  const int hi  = (lane >> 4) & 1;
  int tb = bh / NHEADS, h = bh % NHEADS;
#pragma unroll
  for (int r = 0; r < 8; ++r) {
    int n = n0 + r + (hi << 3);
    int e = e0 + col;
    ua[((size_t)tb * CC + h * DH + e) * NN + n] = acc[r];
  }
}

extern "C" void kernel_launch(void* const* d_in, const int* in_sizes, int n_in,
                              void* d_out, int out_size, void* d_ws, size_t ws_size,
                              hipStream_t stream) {
  (void)in_sizes; (void)n_in; (void)out_size; (void)ws_size;
  const float* x   = (const float*)d_in[0];
  const float* w1  = (const float*)d_in[1];
  const float* dw  = (const float*)d_in[2];
  const float* pw  = (const float*)d_in[3];
  const float* b1w = (const float*)d_in[4];
  const float* b1b = (const float*)d_in[5];
  const float* b1m = (const float*)d_in[6];
  const float* b1v = (const float*)d_in[7];
  const float* b2w = (const float*)d_in[8];
  const float* b2b = (const float*)d_in[9];
  const float* b2m = (const float*)d_in[10];
  const float* b2v = (const float*)d_in[11];
  const float* b3w = (const float*)d_in[12];
  const float* b3b = (const float*)d_in[13];
  const float* b3m = (const float*)d_in[14];
  const float* b3v = (const float*)d_in[15];
  const float* vth = (const float*)d_in[16];
  float* out = (float*)d_out;

  char* p = (char*)d_ws;
  auto alloc = [&](size_t bytes) -> char* {
    char* r = p; p += (bytes + 255) & ~(size_t)255; return r;
  };
  _Float16*      wf16  = (_Float16*)     alloc((size_t)8 * CC * CC * 2);
  float*         coef  = (float*)        alloc((size_t)16 * CC * 4);
  _Float16*      xsT   = (_Float16*)     alloc((size_t)NIMG * NN * CC * 2);
  _Float16*      y1pad = (_Float16*)     alloc((size_t)NIMG * CC * PADAREA * 2);
  _Float16*      y2T   = (_Float16*)     alloc((size_t)NIMG * NN * CC * 2);
  float*         u     = (float*)        alloc((size_t)NIMG * CC * NN * 4);
  _Float16*      q     = (_Float16*)     alloc((size_t)NIMG * CC * NN * 2);
  unsigned char* kT    = (unsigned char*)alloc((size_t)NIMG * CC * NN);
  unsigned char* vT    = (unsigned char*)alloc((size_t)NIMG * CC * NN);
  _Float16*      kvT   = (_Float16*)     alloc((size_t)NIMG * NHEADS * DH * DH * 2);

  cvt_w_kernel<<<4608, 256, 0, stream>>>(w1, pw, wf16);
  bn_prep_kernel<<<6, 256, 0, stream>>>(b1w, b1b, b1m, b1v, b2w, b2b, b2m, b2v,
                                        b3w, b3b, b3m, b3v, coef);
  alif_kernel<<<24576, 256, 0, stream>>>(x, vth, 0, xsT, nullptr, 0);

  const dim3 ggemm(32, 6, NIMG);
  const int gpad = (NIMG * CC * PADAREA + 255) / 256;
  const int gdw  = (NIMG * CC * NN) / 256;

  for (int br = 0; br < 4; ++br) {
    padfill_kernel<<<gpad, 256, 0, stream>>>(coef + (br * 4 + 1) * CC, y1pad);
    conv1x1_wmma_kernel<<<ggemm, 256, 0, stream>>>(
        wf16 + (size_t)br * CC * CC, xsT,
        coef + (br * 4 + 0) * CC, coef + (br * 4 + 1) * CC, y1pad, nullptr, 0);
    dwconv_kernel<<<gdw, 256, 0, stream>>>(y1pad, dw + br * CC * 9, y2T);
    float* gout = (br == 3) ? out : u;
    conv1x1_wmma_kernel<<<ggemm, 256, 0, stream>>>(
        wf16 + (size_t)(4 + br) * CC * CC, y2T,
        coef + (br * 4 + 2) * CC, coef + (br * 4 + 3) * CC, nullptr, gout, 1);

    if (br == 0) alif_kernel<<<24576, 256, 0, stream>>>(u, vth, 1, q, nullptr, 1);
    if (br == 1) alif_kernel<<<24576, 256, 0, stream>>>(u, vth, 2, nullptr, kT, 2);
    if (br == 2) {
      alif_kernel<<<24576, 256, 0, stream>>>(u, vth, 3, nullptr, vT, 2);
      kv_wmma_kernel<<<NIMG * NHEADS, 512, 0, stream>>>(kT, vT, kvT);
      qkv_attn_kernel<<<dim3(128, NIMG * NHEADS), 256, 0, stream>>>(q, kvT, u);
      alif_kernel<<<24576, 256, 0, stream>>>(u, vth, 4, xsT, nullptr, 0);
    }
  }
}
